// SplitResidualVectorQuantizer_37349035606536
// MI455X (gfx1250) — matrix operand order repaired and checked by reading, not compiled, not run on hardware
//
#include <hip/hip_runtime.h>
#include <hip/hip_bf16.h>

// ---------------------------------------------------------------------------
// Types for WMMA fragments (CDNA5 gfx1250, wave32)
// ---------------------------------------------------------------------------
typedef unsigned short u16;
typedef __bf16 bf16;
typedef bf16  v16bf __attribute__((ext_vector_type(16)));
typedef bf16  v8bf  __attribute__((ext_vector_type(8)));
typedef float v8f   __attribute__((ext_vector_type(8)));

#define WPB 8           // waves per block (256 threads)
#define NTOK 16384      // B*T = 4*4096
#define DIM 256         // D
#define CIN 512         // C
#define KCODES 2048     // K
#define NBOOKS 18

#if defined(__has_builtin)
#if __has_builtin(__builtin_amdgcn_sched_barrier)
#define SCHED_FENCE() __builtin_amdgcn_sched_barrier(0)
#endif
#endif
#ifndef SCHED_FENCE
#define SCHED_FENCE()
#endif

__device__ __forceinline__ u16 f2bf(float f) {
    unsigned u = __float_as_uint(f);
    unsigned r = u + 0x7FFFu + ((u >> 16) & 1u);   // round-to-nearest-even
    return (u16)(r >> 16);
}

// A-fragment for v_wmma_f32_16x16x32_bf16:
// lane L holds row M=L&15; kbase = kc*32 + 8*(L>=16); halves at kbase, kbase+16.
__device__ __forceinline__ v16bf load_afrag(const u16* row, int kbase) {
    v8bf lo = *(const v8bf*)(row + kbase);
    v8bf hi = *(const v8bf*)(row + kbase + 16);
    v16bf a;
#pragma unroll
    for (int j = 0; j < 8; ++j) { a[j] = lo[j]; a[8 + j] = hi[j]; }
    return a;
}

__device__ __forceinline__ v8f wmma_bf16(v16bf a, v16bf b, v8f c) {
    return __builtin_amdgcn_wmma_f32_16x16x32_bf16(
        /*neg_a=*/false, a, /*neg_b=*/false, b,
        /*c_mod=*/(short)0, c, /*reuse_a=*/false, /*reuse_b=*/false);
}

// ---------------------------------------------------------------------------
// Pack a row-major f32 matrix src[N][K] into per-lane B-fragment layout:
// dst[((nt*Kc)+kc)*512 + L*16 + j] = bf16(src[nt*16+(L&15)][kc*32+k]),
// k = 8*(L>>4)+j (j<8), 16+8*(L>>4)+(j-8) (j>=8) -> one 32B fragment per lane
// ---------------------------------------------------------------------------
__global__ __launch_bounds__(256) void pack_b_kernel(
    const float* __restrict__ src, u16* __restrict__ dst, int K) {
    int nt = blockIdx.x;
    int Kc = K >> 5;
    for (int kc = 0; kc < Kc; ++kc) {
        for (int e = threadIdx.x; e < 512; e += 256) {
            int L = e >> 4, j = e & 15;
            int kk = (j < 8) ? (8 * (L >> 4) + j) : (16 + 8 * (L >> 4) + (j - 8));
            float v = src[(size_t)(nt * 16 + (L & 15)) * K + kc * 32 + kk];
            dst[((size_t)nt * Kc + kc) * 512 + e] = f2bf(v);
        }
    }
}

// 0.5*||e||^2 per code (argmin dist == argmax(z.e - 0.5||e||^2))
__global__ __launch_bounds__(256) void en2_kernel(
    const float* __restrict__ E, float* __restrict__ en2) {
    __shared__ float red[256];
    int k = blockIdx.x;
    float v = E[(size_t)k * DIM + threadIdx.x];
    red[threadIdx.x] = v * v;
    __syncthreads();
    for (int s = 128; s > 0; s >>= 1) {
        if (threadIdx.x < s) red[threadIdx.x] += red[threadIdx.x + s];
        __syncthreads();
    }
    if (threadIdx.x == 0) en2[k] = 0.5f * red[0];
}

// Transpose x[B][C][T] -> token-major bf16 xtbf[B*T][C] via 16x16 LDS tiles
__global__ __launch_bounds__(256) void xt_kernel(
    const float* __restrict__ x, u16* __restrict__ xtbf) {
    __shared__ u16 tile[16][17];
    int tx = threadIdx.x & 15, ty = threadIdx.x >> 4;
    int id = blockIdx.x;               // B * (C/16) * (T/16) = 4*32*256
    int tt = id % 256; id /= 256;
    int ct = id % 32;  int b = id / 32;
    int c = ct * 16 + ty, t0 = tt * 16;
    tile[ty][tx] = f2bf(x[((size_t)b * CIN + c) * 4096 + t0 + tx]);
    __syncthreads();
    xtbf[((size_t)b * 4096 + t0 + ty) * CIN + ct * 16 + tx] = tile[tx][ty];
}

// ---------------------------------------------------------------------------
// Input projections: xs/xa[tok][d] = sum_c xt[tok][c] * Win[d][c]  (WMMA)
// Fragments loaded in batches of 8 (A and W) with a scheduling fence so the
// loads clause together and the 8 WMMAs run back-to-back.
// ---------------------------------------------------------------------------
__global__ __launch_bounds__(256) void proj_kernel(
    const u16* __restrict__ xtbf, const u16* __restrict__ wsbt,
    const u16* __restrict__ wabt, float* __restrict__ xs_t,
    float* __restrict__ xa_t, float* __restrict__ xa0) {
    int wave = threadIdx.x >> 5, lane = threadIdx.x & 31;
    int tokBase = (blockIdx.x * WPB + wave) * 16;
    int hi = lane >> 4, ln = lane & 15;
    const u16* arow = xtbf + (size_t)(tokBase + ln) * CIN;
    for (int which = 0; which < 2; ++which) {
        const u16* wbt = which ? wabt : wsbt;
        float* dst = which ? xa_t : xs_t;
        for (int nt = 0; nt < 16; ++nt) {           // D/16 output tiles
            v8f acc = {};
#pragma unroll
            for (int h = 0; h < 2; ++h) {           // two batches of 8 kc
                v16bf afh[8], wb[8];
#pragma unroll
                for (int kc = 0; kc < 8; ++kc) {
                    afh[kc] = load_afrag(arow, (h * 8 + kc) * 32 + hi * 8);
                    wb[kc]  = *(const v16bf*)(wbt + ((size_t)nt * 16 + h * 8 + kc) * 512 + lane * 16);
                }
                SCHED_FENCE();
#pragma unroll
                for (int kc = 0; kc < 8; ++kc)
                    acc = wmma_bf16(afh[kc], wb[kc], acc);
                SCHED_FENCE();
            }
#pragma unroll
            for (int i = 0; i < 8; ++i) {
                size_t o = (size_t)(tokBase + i + hi * 8) * DIM + nt * 16 + ln;
                dst[o] = acc[i];
                if (which) xa0[o] = acc[i];
            }
        }
    }
}

// ---------------------------------------------------------------------------
// Fused residual-subtract (previous book) + nearest-code search (this book).
// 2 waves per 16-token tile (each scans half of the 2048 codes); shared
// LDS-staged A tile; full code-tile (8-fragment) register double-buffering
// with sched fences so prefetch loads overlap the 8-WMMA runs.
// ---------------------------------------------------------------------------
__global__ __launch_bounds__(256) void nearest_kernel(
    float* __restrict__ z,              // [NTOK][DIM], updated in place if doSub
    const float* __restrict__ E,        // raw f32 [18][2048][256]
    const u16* __restrict__ ebt,        // packed B-fragments
    const float* __restrict__ en2,      // [18*2048]
    const int* __restrict__ idxPrev, int cbPrev, int cbCur, int doSub,
    int* __restrict__ idxOut) {
    __shared__ __align__(32) u16 ldsZ[4][16 * DIM];    // one tile per wave-pair
    __shared__ float ldsBV[WPB][16];
    __shared__ int   ldsBI[WPB][16];
    int wave = threadIdx.x >> 5, lane = threadIdx.x & 31;
    int pairId = wave >> 1, half = wave & 1;
    int tokBase = (blockIdx.x * 4 + pairId) * 16;
    int hi = lane >> 4, ln = lane & 15;

    // one wave of the pair loads (and optionally updates) the f32 tile,
    // stages the bf16 copy in LDS shared by both waves of the pair
    if (half == 0) {
        for (int m = 0; m < 16; ++m) {
            int tok = tokBase + m;
            float* zrow = z + (size_t)tok * DIM;
            const float* erow =
                doSub ? (E + ((size_t)cbPrev * KCODES + idxPrev[tok]) * DIM) : nullptr;
#pragma unroll
            for (int d = lane; d < DIM; d += 32) {
                float v = zrow[d];
                if (doSub) { v -= erow[d]; zrow[d] = v; }
                ldsZ[pairId][m * DIM + d] = f2bf(v);
            }
        }
    }
    __syncthreads();

    // cache all 8 A-fragments (K=256 -> 8 chunks of 32)
    v16bf af[8];
    const u16* myrow = &ldsZ[pairId][ln * DIM];
#pragma unroll
    for (int kc = 0; kc < 8; ++kc) af[kc] = load_afrag(myrow, kc * 32 + hi * 8);

    float bestV[8]; int bestI[8];
#pragma unroll
    for (int i = 0; i < 8; ++i) { bestV[i] = -3.4e38f; bestI[i] = 0; }

    // this wave scans code tiles nt = half*64 .. half*64+63
    const u16* fbase = ebt + (size_t)cbCur * 128 * 8 * 512 +
                       (size_t)half * 512 * 512 + lane * 16;
    const float* e2base = en2 + cbCur * KCODES;

    v16bf bufs[2][8];
    auto loadNT = [&](int t, v16bf* dstb) {
        const u16* p = fbase + (size_t)t * 8 * 512;
#pragma unroll
        for (int kc = 0; kc < 8; ++kc) dstb[kc] = *(const v16bf*)(p + kc * 512);
    };
    loadNT(0, bufs[0]);
    float e2cur = e2base[(half * 64) * 16 + ln];
#pragma unroll 2
    for (int t = 0; t < 64; ++t) {                  // 64 code tiles per wave
        int cur = t & 1;
        float e2nxt = 0.0f;
        if (t < 63) {                               // prefetch next tile
            loadNT(t + 1, bufs[cur ^ 1]);
            e2nxt = e2base[(half * 64 + t + 1) * 16 + ln];
        }
        SCHED_FENCE();
        v8f acc = {};
#pragma unroll
        for (int kc = 0; kc < 8; ++kc)
            acc = wmma_bf16(af[kc], bufs[cur][kc], acc);
        SCHED_FENCE();
        int code = (half * 64 + t) * 16 + ln;
#pragma unroll
        for (int i = 0; i < 8; ++i) {
            float s = acc[i] - e2cur;
            if (s > bestV[i]) { bestV[i] = s; bestI[i] = code; }
        }
        e2cur = e2nxt;
    }
    // reduce across the 16 lanes of each half-wave (lane = N in C/D layout)
#pragma unroll
    for (int m = 8; m >= 1; m >>= 1) {
#pragma unroll
        for (int i = 0; i < 8; ++i) {
            float ov = __shfl_xor(bestV[i], m, 32);
            int   oi = __shfl_xor(bestI[i], m, 32);
            if (ov > bestV[i] || (ov == bestV[i] && oi < bestI[i])) {
                bestV[i] = ov; bestI[i] = oi;
            }
        }
    }
    if (ln == 0) {
#pragma unroll
        for (int i = 0; i < 8; ++i) {
            ldsBV[wave][i + hi * 8] = bestV[i];
            ldsBI[wave][i + hi * 8] = bestI[i];
        }
    }
    __syncthreads();
    // merge the two halves (half 0 holds lower indices -> wins ties)
    if (half == 0 && ln == 0) {
#pragma unroll
        for (int i = 0; i < 8; ++i) {
            int m = i + hi * 8;
            float v0 = ldsBV[wave][m], v1 = ldsBV[wave + 1][m];
            int   i0 = ldsBI[wave][m], i1 = ldsBI[wave + 1][m];
            idxOut[tokBase + m] = (v1 > v0) ? i1 : i0;
        }
    }
}

// semantic = gather(E[0]); acoustic = xa0 - xa_final + E[17][idxLast] (last subtract)
__global__ __launch_bounds__(256) void semaco_kernel(
    const float* __restrict__ E, const float* __restrict__ xa0,
    const float* __restrict__ xaF, const int* __restrict__ idx0,
    const int* __restrict__ idxLast, u16* __restrict__ sembf,
    u16* __restrict__ acobf) {
    int tok = blockIdx.x, d = threadIdx.x;
    size_t o = (size_t)tok * DIM + d;
    float sem = E[((size_t)idx0[tok]) * DIM + d];
    float aco = xa0[o] - xaF[o] + E[((size_t)17 * KCODES + idxLast[tok]) * DIM + d];
    sembf[o] = f2bf(sem);
    acobf[o] = f2bf(aco);
}

__global__ __launch_bounds__(256) void codes_kernel(
    const int* __restrict__ idxAll, float* __restrict__ outCodes) {
    int i = blockIdx.x * 256 + threadIdx.x;       // over 22*NTOK
    if (i >= 22 * NTOK) return;
    int step = i / NTOK, tok = i % NTOK;
    int b = tok / 4096, t = tok % 4096;
    outCodes[((size_t)b * 22 + step) * 4096 + t] = (float)idxAll[i];
}

// ---------------------------------------------------------------------------
// Decode: out[b][c][t] = sum_d Wout_s[c][d]*sem + Wout_a[c][d]*aco   (WMMA)
// ---------------------------------------------------------------------------
__global__ __launch_bounds__(256) void decode_kernel(
    const u16* __restrict__ sembf, const u16* __restrict__ acobf,
    const u16* __restrict__ wosbt, const u16* __restrict__ woabt,
    float* __restrict__ out) {
    int wave = threadIdx.x >> 5, lane = threadIdx.x & 31;
    int tokBase = (blockIdx.x * WPB + wave) * 16;
    int hi = lane >> 4, ln = lane & 15;
    const u16* srow = sembf + (size_t)(tokBase + ln) * DIM;
    const u16* arow = acobf + (size_t)(tokBase + ln) * DIM;
    // hoist A-fragments for both matrices (reused across all 32 output tiles)
    v16bf af_s[8], af_a[8];
#pragma unroll
    for (int kc = 0; kc < 8; ++kc) {
        af_s[kc] = load_afrag(srow, kc * 32 + hi * 8);
        af_a[kc] = load_afrag(arow, kc * 32 + hi * 8);
    }
    int b = tokBase / 4096, tBase = tokBase % 4096;
    for (int nt = 0; nt < 32; ++nt) {               // C/16 output tiles
        v8f acc = {};
        v16bf wb[8];
#pragma unroll
        for (int kc = 0; kc < 8; ++kc)
            wb[kc] = *(const v16bf*)(wosbt + ((size_t)nt * 8 + kc) * 512 + lane * 16);
        SCHED_FENCE();
#pragma unroll
        for (int kc = 0; kc < 8; ++kc) acc = wmma_bf16(af_s[kc], wb[kc], acc);
        SCHED_FENCE();
#pragma unroll
        for (int kc = 0; kc < 8; ++kc)
            wb[kc] = *(const v16bf*)(woabt + ((size_t)nt * 8 + kc) * 512 + lane * 16);
        SCHED_FENCE();
#pragma unroll
        for (int kc = 0; kc < 8; ++kc) acc = wmma_bf16(af_a[kc], wb[kc], acc);
        SCHED_FENCE();
        int c = nt * 16 + ln;
        float* dst = out + ((size_t)b * CIN + c) * 4096 + tBase + hi * 8;
        *(float4*)(dst)     = make_float4(acc[0], acc[1], acc[2], acc[3]);
        *(float4*)(dst + 4) = make_float4(acc[4], acc[5], acc[6], acc[7]);
    }
}

// ---------------------------------------------------------------------------
extern "C" void kernel_launch(void* const* d_in, const int* in_sizes, int n_in,
                              void* d_out, int out_size, void* d_ws, size_t ws_size,
                              hipStream_t stream) {
    const float* x      = (const float*)d_in[0];
    const float* Win_s  = (const float*)d_in[1];
    const float* Win_a  = (const float*)d_in[2];
    const float* Wout_s = (const float*)d_in[3];
    const float* Wout_a = (const float*)d_in[4];
    const float* E      = (const float*)d_in[5];
    float* out      = (float*)d_out;
    float* outCodes = out + (size_t)4 * CIN * 4096;

    char* p = (char*)d_ws;
    auto carve = [&](size_t bytes) -> void* {
        void* r = (void*)p;
        p += (bytes + 255) & ~(size_t)255;
        return r;
    };
    u16*   ebt    = (u16*)  carve((size_t)NBOOKS * KCODES * DIM * 2);
    u16*   wsbt   = (u16*)  carve((size_t)DIM * CIN * 2);
    u16*   wabt   = (u16*)  carve((size_t)DIM * CIN * 2);
    u16*   wosbt  = (u16*)  carve((size_t)CIN * DIM * 2);
    u16*   woabt  = (u16*)  carve((size_t)CIN * DIM * 2);
    float* en2    = (float*)carve((size_t)NBOOKS * KCODES * 4);
    u16*   xtbf   = (u16*)  carve((size_t)NTOK * CIN * 2);
    float* xs_t   = (float*)carve((size_t)NTOK * DIM * 4);
    float* xa_t   = (float*)carve((size_t)NTOK * DIM * 4);
    float* xa0    = (float*)carve((size_t)NTOK * DIM * 4);
    int*   idxAll = (int*)  carve((size_t)22 * NTOK * 4);
    u16*   sembf  = (u16*)  carve((size_t)NTOK * DIM * 2);
    u16*   acobf  = (u16*)  carve((size_t)NTOK * DIM * 2);

    // one-time packing
    pack_b_kernel<<<(NBOOKS * KCODES) / 16, 256, 0, stream>>>(E, ebt, DIM);
    pack_b_kernel<<<DIM / 16, 256, 0, stream>>>(Win_s, wsbt, CIN);
    pack_b_kernel<<<DIM / 16, 256, 0, stream>>>(Win_a, wabt, CIN);
    pack_b_kernel<<<CIN / 16, 256, 0, stream>>>(Wout_s, wosbt, DIM);
    pack_b_kernel<<<CIN / 16, 256, 0, stream>>>(Wout_a, woabt, DIM);
    en2_kernel<<<NBOOKS * KCODES, 256, 0, stream>>>(E, en2);
    xt_kernel<<<4 * 32 * 256, 256, 0, stream>>>(x, xtbf);

    // encode
    proj_kernel<<<NTOK / 16 / WPB, 256, 0, stream>>>(xtbf, wsbt, wabt, xs_t, xa_t, xa0);

    static const int AC[21] = {1, 2, 3, 4, 5, 6, 7, 8, 9, 10, 11,
                               12, 13, 14, 15, 16, 17, 17, 17, 17, 17};
    const int nb = NTOK / 16 / 4;    // 4 tiles per block (2 waves per tile)
    // step 0: semantic book on xs (no subtract, xs not modified afterwards)
    nearest_kernel<<<nb, 256, 0, stream>>>(xs_t, E, ebt, en2,
                                           (const int*)nullptr, 0, 0, 0, idxAll);
    // step 1: first acoustic book on xa (no prior subtract)
    nearest_kernel<<<nb, 256, 0, stream>>>(xa_t, E, ebt, en2,
                                           (const int*)nullptr, 0, AC[0], 0,
                                           idxAll + NTOK);
    // steps 2..21: subtract previous book's codeword, then search current book
    for (int s = 2; s <= 21; ++s) {
        nearest_kernel<<<nb, 256, 0, stream>>>(xa_t, E, ebt, en2,
                                               idxAll + (size_t)(s - 1) * NTOK,
                                               AC[s - 2], AC[s - 1], 1,
                                               idxAll + (size_t)s * NTOK);
    }

    // decode
    semaco_kernel<<<NTOK, 256, 0, stream>>>(E, xa0, xa_t, idxAll,
                                            idxAll + (size_t)21 * NTOK, sembf, acobf);
    codes_kernel<<<(22 * NTOK + 255) / 256, 256, 0, stream>>>(idxAll, outCodes);
    decode_kernel<<<NTOK / 16 / WPB, 256, 0, stream>>>(sembf, acobf, wosbt, woabt, out);
}
